// CalendarGNNAttn_4252017623143
// MI455X (gfx1250) — compile-verified, matrix operand order checked by reading, not run on hardware
//
#include <hip/hip_runtime.h>
#include <math.h>

// ---------------------------------------------------------------------------
// CalendarGNN-style pipeline for MI455X (gfx1250, wave32, WMMA)
// ---------------------------------------------------------------------------

typedef __attribute__((ext_vector_type(16))) _Float16 v16h;
typedef __attribute__((ext_vector_type(8)))  float    v8f;

#define NUM_S   8192
#define MAX_LEN 64
#define IN_L    1000
#define EMB_L   64
#define HID     128
#define OUTDIM  256

// ---------------- WMMA helpers -------------------------------------------

__device__ __forceinline__ v8f wmma_f16(v16h a, v16h b, v8f c) {
  // D = A(16x32 f16) x B(32x16 f16) + C(16x16 f32)
  return __builtin_amdgcn_wmma_f32_16x16x32_f16(
      /*neg_a=*/false, a, /*neg_b=*/false, b,
      /*c_mod=*/(short)0, c, /*reuse_a=*/false, /*reuse_b=*/false);
}

// Load a 16x32 f16 A-fragment for this lane from an LDS row.
// A layout (ISA 7.12.2, 16-bit A 16x32): lane holds row M=lane%16;
// half j of the v16h maps to k = kbase + (j%8) + (j/8)*16, where
// kbase = kt*32 + ((lane&16)?8:0).  -> two contiguous 16-byte chunks.
__device__ __forceinline__ v16h ldA(const _Float16* rowp, int kbase) {
  v16h a;
  reinterpret_cast<float4*>(&a)[0] = *reinterpret_cast<const float4*>(rowp + kbase);
  reinterpret_cast<float4*>(&a)[1] = *reinterpret_cast<const float4*>(rowp + kbase + 16);
  return a;
}

// Load a pre-packed B-fragment (one 32-byte chunk per lane) from global/L2.
__device__ __forceinline__ v16h ldB(const _Float16* __restrict__ Wp, int fragLinear) {
  const float4* p = reinterpret_cast<const float4*>(Wp) + (size_t)fragLinear * 2;
  v16h b;
  reinterpret_cast<float4*>(&b)[0] = p[0];
  reinterpret_cast<float4*>(&b)[1] = p[1];
  return b;
}

// Scatter a 16x16 f32 C/D fragment into an LDS tile.
// C layout: lane gives N = lane%16; element i is row M = i + ((lane&16)?8:0).
__device__ __forceinline__ void stC(float* g, int ldg, int colBase, v8f acc, int lane) {
  const int n  = lane & 15;
  const int mh = (lane & 16) ? 8 : 0;
#pragma unroll
  for (int i = 0; i < 8; ++i) g[(i + mh) * ldg + colBase + n] = acc[i];
}

// ---------------- small prep kernels -------------------------------------

__global__ void init_kernel(int* maxc) {
  if (threadIdx.x < 4) maxc[threadIdx.x] = 0;
}

__global__ void lens_kernel(const int* __restrict__ vs, int* __restrict__ lens,
                            int S, int T) {
  int s = blockIdx.x * blockDim.x + threadIdx.x;
  if (s < S) {
    int c = 0;
    for (int t = 0; t < T; ++t) c += (vs[s * T + t] > 0) ? 1 : 0;
    lens[s] = c;
  }
}

// Pack W (shape [384][D], N-major) into per-lane-contiguous B fragments:
// Wp[((nt*KT + kt)*32 + lane)*16 + j] = (f16) W[n][k]
// with n = nt*16 + lane%16, k = kt*32 + (j%8) + (j/8)*16 + ((lane&16)?8:0).
__global__ void pack_w(const float* __restrict__ W, _Float16* __restrict__ Wp,
                       int D) {
  const int KT = D / 32;
  const int total = 24 * KT * 32 * 16;
  for (int e = blockIdx.x * blockDim.x + threadIdx.x; e < total;
       e += gridDim.x * blockDim.x) {
    int j = e & 15;
    int lane = (e >> 4) & 31;
    int tile = e >> 9;
    int kt = tile % KT;
    int nt = tile / KT;
    int n = nt * 16 + (lane & 15);
    int k = kt * 32 + (j & 7) + ((j >> 3) << 4) + ((lane & 16) ? 8 : 0);
    Wp[e] = (_Float16)W[(size_t)n * D + k];
  }
}

// brz = bih[0:256]+bhh[0:256] (r,z fused); bin = bih[256:384]; bhn = bhh[256:384]
__global__ void pack_bias(const float* __restrict__ bih,
                          const float* __restrict__ bhh,
                          float* brz, float* bin, float* bhn) {
  int i = threadIdx.x;
  if (i < 256) brz[i] = bih[i] + bhh[i];
  if (i < 128) { bin[i] = bih[256 + i]; bhn[i] = bhh[256 + i]; }
}

// One thread per unit: stable, session-order-preserving grouping.
__global__ void build_groups(const int* __restrict__ src, int stride, int off,
                             int K, int cap, int* __restrict__ counts,
                             int* __restrict__ idx, int* maxc, int S) {
  int u = blockIdx.x * blockDim.x + threadIdx.x;
  if (u >= K) return;
  int c = 0;
  for (int s = 0; s < S; ++s) {
    if (src[s * stride + off] == u) {
      if (c < cap) idx[u * cap + c] = s;
      ++c;
    }
  }
  if (c > cap) c = cap;
  counts[u] = c;
  atomicMax(maxc, c);
}

// ---------------- fused WMMA GRU ------------------------------------------
// One block = 16 logical rows, 8 waves; wave w owns N-tiles {w, w+8, w+16}
// of the 384-wide gate matrix (tiles 0..15 = r|z fused accumulation,
// tiles 16..23 = n gate, kept as separate x / h partial sums).
// Whh B-fragments are register-resident across the T loop (critical path);
// Wih B-fragments stream from L2; x_{t+1} is prefetched into registers
// during the WMMA phase of step t (double-buffered LDS x tile).
// mode 0: item GRU  (x = embs_v[item-1] or 0)
// mode 1: loc GRU   (x = [sess[idx], embs_l[unit]], D=192)
// mode 2: temporal  (x = sess[idx], D=128)
template <int DK>
__global__ void __launch_bounds__(256)
gru_kernel(int mode, int numRows, int T_host, const int* __restrict__ T_dev,
           const int* __restrict__ items, const float* __restrict__ embs_v,
           const int* __restrict__ seqlen, const int* __restrict__ idx, int cap,
           const float* __restrict__ sess, const float* __restrict__ embs_l,
           const _Float16* __restrict__ WihP, const _Float16* __restrict__ WhhP,
           const float* __restrict__ brz, const float* __restrict__ bin,
           const float* __restrict__ bhn, float* __restrict__ h_out) {
  constexpr int D = DK * 32;
  constexpr int PF = (16 * D) / 256;  // prefetch values per thread (8 or 12)
  __shared__ __align__(16) _Float16 xs[2][16][D];
  __shared__ __align__(16) _Float16 hs[16][HID];
  __shared__ float h32[16][HID];
  __shared__ float g_rz[16][256];
  __shared__ float g_in[16][HID];
  __shared__ float g_hn[16][HID];
  __shared__ int Ls[16];

  const int tid = threadIdx.x;
  const int base = blockIdx.x * 16;

  for (int e = tid; e < 16 * HID; e += 256) {
    h32[e >> 7][e & 127] = 0.f;
    hs[e >> 7][e & 127] = (_Float16)0.f;
  }
  if (tid < 16) {
    int row = base + tid;
    Ls[tid] = (row < numRows) ? seqlen[row] : 0;
  }
  __syncthreads();

  int T = (T_host > 0) ? T_host : T_dev[0];
  if (mode != 0 && T > cap) T = cap;

  const int lane = tid & 31;
  const int w = tid >> 5;
  const int mrow = lane & 15;
  const int hiOff = (lane & 16) ? 8 : 0;
  const int nA = w, nB = w + 8, nC = w + 16;

  // gather x_t for the block's 16 rows into per-thread registers
  float xv[PF];
  auto gatherX = [&](int t) {
#pragma unroll
    for (int i = 0; i < PF; ++i) {
      int e = tid + i * 256;
      int m = e / D, d = e - m * D;
      int row = base + m;
      float v = 0.f;
      if (row < numRows && t < Ls[m]) {
        if (mode == 0) {
          int id = items[row * MAX_LEN + t];
          if (id > 0) v = embs_v[(size_t)(id - 1) * 128 + d];
        } else {
          int s = idx[row * cap + t];
          v = (d < 128) ? sess[(size_t)s * HID + d]
                        : embs_l[(size_t)row * EMB_L + (d - 128)];
        }
      }
      xv[i] = v;
    }
  };
  auto storeX = [&](int buf) {
#pragma unroll
    for (int i = 0; i < PF; ++i) {
      int e = tid + i * 256;
      xs[buf][e / D][e % D] = (_Float16)xv[i];
    }
  };

  if (T > 0) {
    gatherX(0);
    storeX(0);
  }

  // Register-resident Whh B-fragments (loop-invariant, on the h critical path)
  v16h bh0[4], bh1[4], bh2[4];
#pragma unroll
  for (int kt = 0; kt < 4; ++kt) {
    bh0[kt] = ldB(WhhP, (nA * 4 + kt) * 32 + lane);
    bh1[kt] = ldB(WhhP, (nB * 4 + kt) * 32 + lane);
    bh2[kt] = ldB(WhhP, (nC * 4 + kt) * 32 + lane);
  }
  __syncthreads();

  for (int t = 0; t < T; ++t) {
    const int cur = t & 1;
    const bool haveNext = (t + 1 < T);
    if (haveNext) gatherX(t + 1);  // overlap gather with WMMA phase

    // ---- WMMA: gi = x@Wih^T, gh = h@Whh^T (r,z fused; n split) ----------
    {
      const _Float16* xr = &xs[cur][mrow][0];
      const _Float16* hr = &hs[mrow][0];
      v8f accA = {}, accB = {};
#pragma unroll
      for (int kt = 0; kt < DK; ++kt) {
        v16h a = ldA(xr, kt * 32 + hiOff);
        accA = wmma_f16(a, ldB(WihP, (nA * DK + kt) * 32 + lane), accA);
        accB = wmma_f16(a, ldB(WihP, (nB * DK + kt) * 32 + lane), accB);
      }
#pragma unroll
      for (int kt = 0; kt < 4; ++kt) {
        v16h a = ldA(hr, kt * 32 + hiOff);
        accA = wmma_f16(a, bh0[kt], accA);
        accB = wmma_f16(a, bh1[kt], accB);
      }
      stC(&g_rz[0][0], 256, nA * 16, accA, lane);
      stC(&g_rz[0][0], 256, nB * 16, accB, lane);

      v8f accI = {}, accH = {};
#pragma unroll
      for (int kt = 0; kt < DK; ++kt) {
        v16h a = ldA(xr, kt * 32 + hiOff);
        accI = wmma_f16(a, ldB(WihP, (nC * DK + kt) * 32 + lane), accI);
      }
#pragma unroll
      for (int kt = 0; kt < 4; ++kt) {
        v16h a = ldA(hr, kt * 32 + hiOff);
        accH = wmma_f16(a, bh2[kt], accH);
      }
      stC(&g_in[0][0], HID, (nC - 16) * 16, accI, lane);
      stC(&g_hn[0][0], HID, (nC - 16) * 16, accH, lane);
    }
    if (haveNext) storeX(cur ^ 1);
    __syncthreads();

    // ---- gates + masked hidden update (fp32) ----------------------------
    for (int e = tid; e < 16 * HID; e += 256) {
      int m = e >> 7, k = e & 127;
      float r = 1.f / (1.f + __expf(-(g_rz[m][k] + brz[k])));
      float z = 1.f / (1.f + __expf(-(g_rz[m][k + 128] + brz[128 + k])));
      float n = tanhf(g_in[m][k] + bin[k] + r * (g_hn[m][k] + bhn[k]));
      float hp = h32[m][k];
      bool act = ((base + m) < numRows) && (t < Ls[m]);
      float hv = act ? ((1.f - z) * n + z * hp) : hp;
      h32[m][k] = hv;
      hs[m][k] = (_Float16)hv;
    }
    __syncthreads();
  }

  for (int e = tid; e < 16 * HID; e += 256) {
    int m = e >> 7, k = e & 127;
    int row = base + m;
    if (row < numRows) h_out[(size_t)row * HID + k] = h32[m][k];
  }
}

// ---------------- attention + FC ------------------------------------------

__global__ void attend_kernel(const float* __restrict__ temp,
                              const int* __restrict__ tcnt, int Kt,
                              const float* __restrict__ spat,
                              const int* __restrict__ scnt, int Ks,
                              const float* __restrict__ Wt,
                              const float* __restrict__ bt,
                              const float* __restrict__ Ws,
                              const float* __restrict__ bs,
                              float* __restrict__ pat_out /* [256] */) {
  __shared__ float q[HID];
  __shared__ float qW[HID];
  __shared__ float sc[1024];
  __shared__ float nsv, ntv;
  const int tid = threadIdx.x;

  if (tid == 0) { int c = 0; for (int u = 0; u < Ks; ++u) c += (scnt[u] > 0); nsv = fmaxf((float)c, 1.f); }
  if (tid == 1) { int c = 0; for (int i = 0; i < Kt; ++i) c += (tcnt[i] > 0); ntv = fmaxf((float)c, 1.f); }
  __syncthreads();

  // spat_q = masked mean of spat
  if (tid < HID) {
    float s = 0.f;
    for (int u = 0; u < Ks; ++u)
      if (scnt[u] > 0) s += spat[(size_t)u * HID + tid];
    q[tid] = s / nsv;
  }
  __syncthreads();
  // qW = spat_q @ Wt
  if (tid < HID) {
    float s = 0.f;
    for (int d = 0; d < HID; ++d) s += q[d] * Wt[(size_t)d * HID + tid];
    qW[tid] = s;
  }
  __syncthreads();
  // scores over temp units, masked softmax
  for (int i = tid; i < Kt; i += 256) {
    float s = bt[0];
    for (int d = 0; d < HID; ++d) s += qW[d] * temp[(size_t)i * HID + d];
    sc[i] = (tcnt[i] > 0) ? s : -1e9f;
  }
  __syncthreads();
  if (tid == 0) {
    float mx = -3.4e38f;
    for (int i = 0; i < Kt; ++i) mx = fmaxf(mx, sc[i]);
    float sum = 0.f;
    for (int i = 0; i < Kt; ++i) { sc[i] = __expf(sc[i] - mx); sum += sc[i]; }
    for (int i = 0; i < Kt; ++i) sc[i] /= sum;
  }
  __syncthreads();
  if (tid < HID) {
    float s = 0.f;
    for (int i = 0; i < Kt; ++i) s += sc[i] * temp[(size_t)i * HID + tid];
    pat_out[tid] = s;
  }
  __syncthreads();
  // temp_q = masked mean of temp
  if (tid < HID) {
    float s = 0.f;
    for (int i = 0; i < Kt; ++i)
      if (tcnt[i] > 0) s += temp[(size_t)i * HID + tid];
    q[tid] = s / ntv;
  }
  __syncthreads();
  if (tid < HID) {
    float s = 0.f;
    for (int d = 0; d < HID; ++d) s += q[d] * Ws[(size_t)d * HID + tid];
    qW[tid] = s;
  }
  __syncthreads();
  for (int u = tid; u < Ks; u += 256) {
    float s = bs[0];
    for (int d = 0; d < HID; ++d) s += qW[d] * spat[(size_t)u * HID + d];
    sc[u] = (scnt[u] > 0) ? s : -1e9f;
  }
  __syncthreads();
  if (tid == 0) {
    float mx = -3.4e38f;
    for (int u = 0; u < Ks; ++u) mx = fmaxf(mx, sc[u]);
    float sum = 0.f;
    for (int u = 0; u < Ks; ++u) { sc[u] = __expf(sc[u] - mx); sum += sc[u]; }
    for (int u = 0; u < Ks; ++u) sc[u] /= sum;
  }
  __syncthreads();
  if (tid < HID) {
    float s = 0.f;
    for (int u = 0; u < Ks; ++u) s += sc[u] * spat[(size_t)u * HID + tid];
    pat_out[HID + tid] = s;
  }
}

__global__ void fc_kernel(const float* __restrict__ user,
                          const float* __restrict__ W,
                          const float* __restrict__ b, float* __restrict__ out) {
  int o = blockIdx.x * blockDim.x + threadIdx.x;
  if (o < OUTDIM) {
    float s = b[o];
    for (int j = 0; j < 6 * HID; ++j) s += user[j] * W[(size_t)o * 6 * HID + j];
    out[o] = s;
  }
}

// ---------------- launch ---------------------------------------------------

extern "C" void kernel_launch(void* const* d_in, const int* in_sizes, int n_in,
                              void* d_out, int out_size, void* d_ws,
                              size_t ws_size, hipStream_t stream) {
  (void)in_sizes; (void)n_in; (void)out_size; (void)ws_size;

  // --- inputs, flattened in setup_inputs() insertion order ---
  const int*   u_s_vs = (const int*)d_in[0];     // [8192*64]
  const int*   u_s_ts = (const int*)d_in[1];     // [8192*4] (day,hour,week,wday)
  const int*   u_s_l  = (const int*)d_in[2];     // [8192]
  const float* embs_v = (const float*)d_in[3];   // [100000*128]
  const float* embs_l = (const float*)d_in[4];   // [1000*64]
  // grus: item(5..8), h(9..12), w(13..16), y(17..20), l(21..24) = Wih,Whh,bih,bhh
  const float* gW[5][4];
  for (int g = 0; g < 5; ++g)
    for (int k = 0; k < 4; ++k) gW[g][k] = (const float*)d_in[5 + g * 4 + k];
  // bil: hpat_l, lpat_h, wpat_l, lpat_w, ypat_l, lpat_y -> W,b pairs at 25..36
  const float* bilW[6]; const float* bilB[6];
  for (int i = 0; i < 6; ++i) {
    bilW[i] = (const float*)d_in[25 + 2 * i];
    bilB[i] = (const float*)d_in[26 + 2 * i];
  }
  const float* fc_W = (const float*)d_in[37];
  const float* fc_b = (const float*)d_in[38];
  float* out = (float*)d_out;

  // --- workspace bump allocator ---
  char* p = (char*)d_ws;
  auto alloc = [&](size_t bytes) -> void* {
    void* r = (void*)p;
    p += (bytes + 255) & ~(size_t)255;
    return r;
  };
  float*    sess  = (float*)alloc((size_t)NUM_S * HID * 4);
  int*      lens  = (int*)alloc((size_t)NUM_S * 4);
  int*      maxc  = (int*)alloc(4 * 4);
  // packed f16 weights + biases per GRU (order: item, h, w, y, l)
  _Float16* WihP[5]; _Float16* WhhP[5];
  float*    brz[5]; float* bin[5]; float* bhn[5];
  const int Din[5] = {128, 128, 128, 128, 192};
  for (int g = 0; g < 5; ++g) {
    WihP[g] = (_Float16*)alloc((size_t)24 * (Din[g] / 32) * 512 * 2);
    WhhP[g] = (_Float16*)alloc((size_t)24 * 4 * 512 * 2);
    brz[g] = (float*)alloc(256 * 4);
    bin[g] = (float*)alloc(128 * 4);
    bhn[g] = (float*)alloc(128 * 4);
  }
  const int CAP_L = 1024, CAP_T = 2048, CAP_Y = 8192;
  int* cnt_l = (int*)alloc(IN_L * 4);
  int* idx_l = (int*)alloc((size_t)IN_L * CAP_L * 4);
  int* cnt_h = (int*)alloc(24 * 4);
  int* idx_h = (int*)alloc((size_t)24 * CAP_T * 4);
  int* cnt_w = (int*)alloc(53 * 4);
  int* idx_w = (int*)alloc((size_t)53 * CAP_T * 4);
  int* cnt_y = (int*)alloc(7 * 4);
  int* idx_y = (int*)alloc((size_t)7 * CAP_Y * 4);
  float* spat = (float*)alloc((size_t)IN_L * HID * 4);
  float* hh   = (float*)alloc((size_t)24 * HID * 4);
  float* hw   = (float*)alloc((size_t)53 * HID * 4);
  float* hy   = (float*)alloc((size_t)7 * HID * 4);
  float* pats = (float*)alloc(6 * HID * 4);

  // --- prep ---
  init_kernel<<<1, 32, 0, stream>>>(maxc);
  lens_kernel<<<(NUM_S + 255) / 256, 256, 0, stream>>>(u_s_vs, lens, NUM_S, MAX_LEN);
  for (int g = 0; g < 5; ++g) {
    int totIh = 24 * (Din[g] / 32) * 512;
    pack_w<<<(totIh + 255) / 256, 256, 0, stream>>>(gW[g][0], WihP[g], Din[g]);
    pack_w<<<(24 * 4 * 512 + 255) / 256, 256, 0, stream>>>(gW[g][1], WhhP[g], 128);
    pack_bias<<<1, 256, 0, stream>>>(gW[g][2], gW[g][3], brz[g], bin[g], bhn[g]);
  }

  // --- stage 1: item -> session GRU (the heavy one) ---
  gru_kernel<4><<<(NUM_S + 15) / 16, 256, 0, stream>>>(
      0, NUM_S, MAX_LEN, (const int*)nullptr, u_s_vs, embs_v, lens,
      (const int*)nullptr, 1, (const float*)nullptr, (const float*)nullptr,
      WihP[0], WhhP[0], brz[0], bin[0], bhn[0], sess);

  // --- stage 2: grouping ---
  build_groups<<<(IN_L + 127) / 128, 128, 0, stream>>>(u_s_l, 1, 0, IN_L, CAP_L,
                                                       cnt_l, idx_l, &maxc[0], NUM_S);
  build_groups<<<1, 64, 0, stream>>>(u_s_ts, 4, 1, 24, CAP_T, cnt_h, idx_h, &maxc[1], NUM_S);
  build_groups<<<1, 64, 0, stream>>>(u_s_ts, 4, 2, 53, CAP_T, cnt_w, idx_w, &maxc[2], NUM_S);
  build_groups<<<1, 32, 0, stream>>>(u_s_ts, 4, 3, 7, CAP_Y, cnt_y, idx_y, &maxc[3], NUM_S);

  // --- stage 3: unit GRUs (device-side trip count = maxc) ---
  gru_kernel<6><<<(IN_L + 15) / 16, 256, 0, stream>>>(
      1, IN_L, -1, &maxc[0], (const int*)nullptr, (const float*)nullptr, cnt_l,
      idx_l, CAP_L, sess, embs_l, WihP[4], WhhP[4], brz[4], bin[4], bhn[4], spat);
  gru_kernel<4><<<2, 256, 0, stream>>>(
      2, 24, -1, &maxc[1], (const int*)nullptr, (const float*)nullptr, cnt_h,
      idx_h, CAP_T, sess, (const float*)nullptr,
      WihP[1], WhhP[1], brz[1], bin[1], bhn[1], hh);
  gru_kernel<4><<<4, 256, 0, stream>>>(
      2, 53, -1, &maxc[2], (const int*)nullptr, (const float*)nullptr, cnt_w,
      idx_w, CAP_T, sess, (const float*)nullptr,
      WihP[2], WhhP[2], brz[2], bin[2], bhn[2], hw);
  gru_kernel<4><<<1, 256, 0, stream>>>(
      2, 7, -1, &maxc[3], (const int*)nullptr, (const float*)nullptr, cnt_y,
      idx_y, CAP_Y, sess, (const float*)nullptr,
      WihP[3], WhhP[3], brz[3], bin[3], bhn[3], hy);

  // --- stage 4: interactive attention (hour, week, weekday) + FC ---
  attend_kernel<<<1, 256, 0, stream>>>(hh, cnt_h, 24, spat, cnt_l, IN_L,
                                       bilW[0], bilB[0], bilW[1], bilB[1], pats);
  attend_kernel<<<1, 256, 0, stream>>>(hw, cnt_w, 53, spat, cnt_l, IN_L,
                                       bilW[2], bilB[2], bilW[3], bilB[3], pats + 256);
  attend_kernel<<<1, 256, 0, stream>>>(hy, cnt_y, 7, spat, cnt_l, IN_L,
                                       bilW[4], bilB[4], bilW[5], bilB[5], pats + 512);
  fc_kernel<<<1, 256, 0, stream>>>(pats, fc_W, fc_b, out);
}
